// MultiHeadAttention_90855738180367
// MI455X (gfx1250) — matrix operand order repaired and checked by reading, not compiled, not run on hardware
//
#include <hip/hip_runtime.h>
#include <hip/hip_bf16.h>

// MHA forward for B=4, L=2048, D=512, H=8, HD=64 on gfx1250 (wave32, WMMA).
// Pipeline: [cvt weights f32->f16] -> [Q/K/V projection GEMMs (WMMA, weights
// staged to LDS via async-to-LDS)] -> [flash attention: K/V chunks double-
// buffered in LDS via global_load_async_to_lds_b128, S^T-form online softmax,
// all matmuls via v_wmma_f32_16x16x32_f16] -> [output projection -> f32].

#define BATCH 4
#define SEQ   2048
#define DMODEL 512
#define NHEAD 8
#define HDIM  64
#define MROWS (BATCH * SEQ)          // 8192
#define DD    (DMODEL * DMODEL)      // 262144
#define NBHLD ((size_t)BATCH * NHEAD * SEQ * HDIM)  // 4194304 halves

typedef __attribute__((ext_vector_type(16))) _Float16 v16h;
typedef __attribute__((ext_vector_type(8)))  _Float16 v8h;
typedef __attribute__((ext_vector_type(8)))  float    v8f;

union H16 { v16h v; v8h h[2]; };

__device__ __forceinline__ v8f wmma_f16(v16h a, v16h b, v8f c) {
  // D = A(16x32) * B(32x16) + C, f32 accumulate
  return __builtin_amdgcn_wmma_f32_16x16x32_f16(
      false, a, false, b, (short)0, c, false, false);
}

// Async copy 16B/lane from global to LDS (ASYNCcnt-tracked).
// LDS destination is the vdst VGPR plus the offset (ISA 15.18.3), vaddr = 64-bit GV.
__device__ __forceinline__ void async_b128(unsigned lds_off, const void* g) {
  asm volatile("global_load_async_to_lds_b128 %0, %1, off"
               :: "v"(lds_off), "v"((unsigned long long)(uintptr_t)g)
               : "memory");
}
__device__ __forceinline__ void wait_async_le2() {
  asm volatile("s_wait_asynccnt 0x2" ::: "memory");
}
__device__ __forceinline__ void wait_async_0() {
  asm volatile("s_wait_asynccnt 0x0" ::: "memory");
}
__device__ __forceinline__ unsigned lds_off_of(const void* p) {
  return (unsigned)(uintptr_t)p;   // low 32 bits of flat LDS address
}

// ---- Fragment loaders (contiguous k order k = 16h + s, same for A and B) ----------------------
// A-matrix 16x32 f16: lane<16 -> row M=lane, elems[0..15]=K k0..k0+15 ;
// lane>=16 -> row M=lane-16, K k0+16..k0+31
__device__ __forceinline__ v16h load_afrag(const _Float16* src, int row0,
                                           int stride, int k0) {
  int lane = threadIdx.x & 31;
  int r = lane & 15, hi = lane >> 4;
  const _Float16* p = src + (size_t)(row0 + r) * stride + k0 + hi * 16;
  H16 u;
  u.h[0] = *(const v8h*)(p);
  u.h[1] = *(const v8h*)(p + 8);
  return u.v;
}

// Same A layout but source is f32 (convert on the fly)
__device__ __forceinline__ v16h load_afrag_f32(const float* src, int row0,
                                               int stride, int k0) {
  int lane = threadIdx.x & 31;
  int r = lane & 15, hi = lane >> 4;
  const float* p = src + (size_t)(row0 + r) * stride + k0 + hi * 16;
  v16h out;
#pragma unroll
  for (int i = 0; i < 8; ++i) {
    out[i]     = (_Float16)p[i];
    out[8 + i] = (_Float16)p[8 + i];
  }
  return out;
}

// B-matrix 32x16 f16, where B = src^T (src row-major [N, stride]):
// lane<16 -> col N=lane, elems[0..15] = K k0..k0+15 (contiguous in src row);
// lane>=16 -> col N=lane-16, K k0+16..k0+31.
__device__ __forceinline__ v16h load_bfrag(const _Float16* src, int col0,
                                           int stride, int k0) {
  int lane = threadIdx.x & 31;
  int n = lane & 15, hi = lane >> 4;
  const _Float16* p = src + (size_t)(col0 + n) * stride + k0 + hi * 16;
  H16 u;
  u.h[0] = *(const v8h*)(p);
  u.h[1] = *(const v8h*)(p + 8);
  return u.v;
}

// ---- Weight f32 -> f16 ----------------------------------------------------
__global__ void cvt_f32_f16(const float* __restrict__ src,
                            _Float16* __restrict__ dst, int n) {
  int i = blockIdx.x * blockDim.x + threadIdx.x;
  if (i < n) dst[i] = (_Float16)src[i];
}

// ---- Projection GEMM: out = relu(X @ W^T + b) -----------------------------
// MODE 0/1: store f16 to [B,H,L,HD] (Q,K)   MODE 2: f16 transposed [B,H,HD,L]
// MODE 3: store f32 row-major [M, 512] to d_out.
// Grid: (MROWS/128, 512/64). Block 256 = 8 waves; wave -> 16x64 output tile.
// The 64x512 weight tile (64 KB) is async-copied to LDS once per block; the
// A fragment is software-pipelined one k-step ahead of the 4 WMMAs.
template <int MODE, typename TIn>
__global__ void __launch_bounds__(256)
proj_kernel(const TIn* __restrict__ X, const _Float16* __restrict__ W,
            const float* __restrict__ bias, _Float16* __restrict__ dsth,
            float* __restrict__ dstf) {
  __shared__ _Float16 Wl[64 * DMODEL];   // 64 KB
  int tid = threadIdx.x;
  int lane = tid & 31;
  int wave = tid >> 5;
  int m0 = blockIdx.x * 128 + wave * 16;
  int n0 = blockIdx.y * 64;

  // Stage W[n0..n0+63][0..511] (contiguous 64 KB) into LDS: 16 x b128/thread.
  {
    const char* ws = (const char*)(W + (size_t)n0 * DMODEL);
    unsigned wl = lds_off_of(&Wl[0]);
#pragma unroll
    for (int i = 0; i < 16; ++i) {
      int off = (tid + i * 256) * 16;
      async_b128(wl + off, ws + off);
    }
    wait_async_0();
  }
  __syncthreads();

  v8f acc[4] = {v8f{}, v8f{}, v8f{}, v8f{}};
  v16h a_cur;
  if constexpr (sizeof(TIn) == 4)
    a_cur = load_afrag_f32((const float*)X, m0, DMODEL, 0);
  else
    a_cur = load_afrag((const _Float16*)X, m0, DMODEL, 0);
  for (int k0 = 0; k0 < DMODEL; k0 += 32) {
    v16h a_nxt = a_cur;
    if (k0 + 32 < DMODEL) {
      if constexpr (sizeof(TIn) == 4)
        a_nxt = load_afrag_f32((const float*)X, m0, DMODEL, k0 + 32);
      else
        a_nxt = load_afrag((const _Float16*)X, m0, DMODEL, k0 + 32);
    }
#pragma unroll
    for (int nt = 0; nt < 4; ++nt) {
      v16h bf = load_bfrag(Wl, nt * 16, DMODEL, k0);
      acc[nt] = wmma_f16(a_cur, bf, acc[nt]);
    }
    a_cur = a_nxt;
  }

  int hiHalf = lane >> 4;          // C layout: lane>=16 -> rows M+8
  int ncol_l = lane & 15;
#pragma unroll
  for (int nt = 0; nt < 4; ++nt) {
    int col = n0 + nt * 16 + ncol_l;
    float bv = bias[col];
    if constexpr (MODE == 0 || MODE == 1) {
      int h_ = col >> 6, hd_ = col & 63;
#pragma unroll
      for (int e = 0; e < 8; ++e) {
        int m = m0 + e + hiHalf * 8;
        int b_ = m >> 11, l_ = m & 2047;
        float val = fmaxf(acc[nt][e] + bv, 0.0f);
        dsth[((((size_t)b_ * NHEAD + h_) * SEQ + l_) * HDIM) + hd_] =
            (_Float16)val;
      }
    } else if constexpr (MODE == 2) {
      int h_ = col >> 6, hd_ = col & 63;
      int b_ = m0 >> 11;
      int l0 = (m0 & 2047) + hiHalf * 8;   // 8 consecutive l values
      v8h pk;
#pragma unroll
      for (int e = 0; e < 8; ++e)
        pk[e] = (_Float16)fmaxf(acc[nt][e] + bv, 0.0f);
      *(v8h*)(dsth + (((size_t)b_ * NHEAD + h_) * HDIM + hd_) * SEQ + l0) = pk;
    } else {  // MODE 3: f32 output
#pragma unroll
      for (int e = 0; e < 8; ++e) {
        int m = m0 + e + hiHalf * 8;
        dstf[(size_t)m * DMODEL + col] = fmaxf(acc[nt][e] + bv, 0.0f);
      }
    }
  }
  (void)dsth; (void)dstf;
}

// ---- Flash attention ------------------------------------------------------
// Block = one (b,h) with 128 queries (8 waves x 16). K/V 32-key chunks are
// double-buffered in LDS via async-to-LDS (shared by all 8 waves). Each wave
// computes S^T = K * Q^T so the C layout maps lane -> query: online-softmax
// state is a per-lane scalar (one shfl_xor(16) merge). O^T = V^T * P^T.
__global__ void __launch_bounds__(256)
flash_kernel(const _Float16* __restrict__ Qh, const _Float16* __restrict__ Kh,
             const _Float16* __restrict__ Vt, _Float16* __restrict__ attn16) {
  __shared__ _Float16 Klds[2][32 * HDIM];   // 2 x 4 KB
  __shared__ _Float16 Vlds[2][HDIM * 32];   // 2 x 4 KB

  int tid = threadIdx.x;
  int lane = tid & 31;
  int wave = tid >> 5;
  int qb = blockIdx.x & 15;          // 16 query-blocks of 128
  int h  = (blockIdx.x >> 4) & 7;
  int b  = blockIdx.x >> 7;

  const _Float16* Qb = Qh + ((size_t)b * NHEAD + h) * SEQ * HDIM;
  const _Float16* Kb = Kh + ((size_t)b * NHEAD + h) * SEQ * HDIM;
  const _Float16* Vb = Vt + ((size_t)b * NHEAD + h) * HDIM * SEQ;

  int q0 = qb * 128 + wave * 16;
  // Q^T B-fragments (reused across all key chunks)
  v16h bq0 = load_bfrag(Qb, q0, HDIM, 0);
  v16h bq1 = load_bfrag(Qb, q0, HDIM, 32);

  // Per-thread async-copy addressing: K chunk is 4 KB contiguous; V chunk is
  // 64 rows (hd) x 64 B with row stride SEQ halves.
  unsigned klo[2] = {lds_off_of(&Klds[0][0]) + (unsigned)tid * 16,
                     lds_off_of(&Klds[1][0]) + (unsigned)tid * 16};
  int vrow = tid >> 2;               // hd row 0..63
  int vsub = (tid & 3) * 16;         // byte offset within 64 B row-chunk
  unsigned vlo[2] = {lds_off_of(&Vlds[0][0]) + (unsigned)(vrow * 64 + vsub),
                     lds_off_of(&Vlds[1][0]) + (unsigned)(vrow * 64 + vsub)};
  const char* kg = (const char*)Kb;
  const char* vg = (const char*)(Vb + (size_t)vrow * SEQ) + vsub;

  // Prologue: stream chunk 0 into buffer 0.
  async_b128(klo[0], kg + (size_t)tid * 16);
  async_b128(vlo[0], vg);

  v8f o0 = {}, o1 = {}, o2 = {}, o3 = {};
  float m2 = -1e30f, rs = 0.0f;
  const float sc = 0.044194173824159216f * 1.4426950408889634f; // 1/sqrt(512)*log2e
  bool hiL = lane >= 16;

  for (int kb = 0; kb < SEQ; kb += 32) {
    int cur = (kb >> 5) & 1;
    if (kb + 32 < SEQ) {
      int nb = cur ^ 1;
      async_b128(klo[nb], kg + (size_t)(kb + 32) * 128 + (size_t)tid * 16);
      async_b128(vlo[nb], vg + (size_t)(kb + 32) * 2);
      wait_async_le2();              // current buffer's copies complete
    } else {
      wait_async_0();
    }
    __syncthreads();

    const _Float16* Kc = &Klds[cur][0];
    const _Float16* Vc = &Vlds[cur][0];

    // S^T tiles: rows = keys (local 0..31), cols = queries
    v16h a0 = load_afrag(Kc, 0, HDIM, 0);
    v16h a1 = load_afrag(Kc, 0, HDIM, 32);
    v8f s0 = {};
    s0 = wmma_f16(a0, bq0, s0);
    s0 = wmma_f16(a1, bq1, s0);
    v16h a2 = load_afrag(Kc, 16, HDIM, 0);
    v16h a3 = load_afrag(Kc, 16, HDIM, 32);
    v8f s1 = {};
    s1 = wmma_f16(a2, bq0, s1);
    s1 = wmma_f16(a3, bq1, s1);

    // Online softmax (per-lane = per-query; merge half-wave twin via xor 16)
    float p0[8], p1[8];
    float lm = -1e30f;
#pragma unroll
    for (int e = 0; e < 8; ++e) {
      p0[e] = s0[e] * sc;
      p1[e] = s1[e] * sc;
      lm = fmaxf(lm, fmaxf(p0[e], p1[e]));
    }
    lm = fmaxf(lm, __shfl_xor(lm, 16, 32));
    float mn = fmaxf(m2, lm);
    float alpha = exp2f(m2 - mn);
    m2 = mn;
    float lsum = 0.0f;
#pragma unroll
    for (int e = 0; e < 8; ++e) {
      p0[e] = exp2f(p0[e] - mn);
      p1[e] = exp2f(p1[e] - mn);
      lsum += p0[e] + p1[e];
    }
    lsum += __shfl_xor(lsum, 16, 32);
    rs = rs * alpha + lsum;
#pragma unroll
    for (int e = 0; e < 8; ++e) {
      o0[e] *= alpha; o1[e] *= alpha; o2[e] *= alpha; o3[e] *= alpha;
    }

    // Build P^T B-fragment (32 keys x 16 queries) in registers.
    v16h pf;
#pragma unroll
    for (int e = 0; e < 8; ++e) {
      float x0 = __shfl_xor(p0[e], 16, 32);
      float x1 = __shfl_xor(p1[e], 16, 32);
      pf[e]     = (_Float16)(hiL ? x1 : p0[e]);
      pf[8 + e] = (_Float16)(hiL ? p1[e] : x0);
    }

    // O^T += V^T * P^T  (4 tiles over HD=64)
    v16h av;
    av = load_afrag(Vc, 0,  32, 0); o0 = wmma_f16(av, pf, o0);
    av = load_afrag(Vc, 16, 32, 0); o1 = wmma_f16(av, pf, o1);
    av = load_afrag(Vc, 32, 32, 0); o2 = wmma_f16(av, pf, o2);
    av = load_afrag(Vc, 48, 32, 0); o3 = wmma_f16(av, pf, o3);

    __syncthreads();   // all waves done reading before buffer reuse
  }

  float inv = 1.0f / rs;
  int qrow = b * SEQ + q0 + (lane & 15);
  _Float16* dst = attn16 + (size_t)qrow * DMODEL + h * HDIM + (hiL ? 8 : 0);
  v8f* oacc[4] = {&o0, &o1, &o2, &o3};
#pragma unroll
  for (int t = 0; t < 4; ++t) {
    v8h pk;
#pragma unroll
    for (int e = 0; e < 8; ++e) pk[e] = (_Float16)((*oacc[t])[e] * inv);
    *(v8h*)(dst + t * 16) = pk;
  }
}

// ---- Host-side launch -----------------------------------------------------
extern "C" void kernel_launch(void* const* d_in, const int* in_sizes, int n_in,
                              void* d_out, int out_size, void* d_ws,
                              size_t ws_size, hipStream_t stream) {
  const float* q  = (const float*)d_in[0];
  const float* k  = (const float*)d_in[1];
  const float* v  = (const float*)d_in[2];
  const float* Wq = (const float*)d_in[3];
  const float* bq = (const float*)d_in[4];
  const float* Wk = (const float*)d_in[5];
  const float* bk = (const float*)d_in[6];
  const float* Wv = (const float*)d_in[7];
  const float* bv = (const float*)d_in[8];
  const float* Wo = (const float*)d_in[9];
  const float* bo = (const float*)d_in[10];
  // d_in[11], d_in[12] (rel tables) are dead in the reference.
  float* out = (float*)d_out;

  _Float16* ws = (_Float16*)d_ws;
  _Float16* W16q = ws;                  // 4 x 512*512 halves = 2 MB
  _Float16* W16k = W16q + DD;
  _Float16* W16v = W16k + DD;
  _Float16* W16o = W16v + DD;
  _Float16* Qh   = W16o + DD;           // [B,H,L,64]  8 MB
  _Float16* Khd  = Qh + NBHLD;          // [B,H,L,64]  8 MB
  _Float16* Vt   = Khd + NBHLD;         // [B,H,64,L]  8 MB
  _Float16* attn = Vt + NBHLD;          // [B*L, 512]  8 MB

  // 1) Convert weights to f16
  int cg = (DD + 255) / 256;
  cvt_f32_f16<<<cg, 256, 0, stream>>>(Wq, W16q, DD);
  cvt_f32_f16<<<cg, 256, 0, stream>>>(Wk, W16k, DD);
  cvt_f32_f16<<<cg, 256, 0, stream>>>(Wv, W16v, DD);
  cvt_f32_f16<<<cg, 256, 0, stream>>>(Wo, W16o, DD);

  // 2) Q/K/V projections (relu(X @ W^T + b)), head-split stores
  dim3 pb(256), pg(MROWS / 128, DMODEL / 64);
  proj_kernel<0, float><<<pg, pb, 0, stream>>>(q, W16q, bq, Qh, nullptr);
  proj_kernel<1, float><<<pg, pb, 0, stream>>>(k, W16k, bk, Khd, nullptr);
  proj_kernel<2, float><<<pg, pb, 0, stream>>>(v, W16v, bv, Vt, nullptr);

  // 3) Flash attention: 512 blocks x (one (b,h), 128 queries)
  flash_kernel<<<512, 256, 0, stream>>>(Qh, Khd, Vt, attn);

  // 4) Output projection -> f32
  proj_kernel<3, _Float16><<<pg, pb, 0, stream>>>(attn, W16o, bo, nullptr, out);
}
